// A2aSparseMLP_65833258713873
// MI455X (gfx1250) — compile-verified, matrix operand order checked by reading, not run on hardware
//
#include <hip/hip_runtime.h>
#include <hip/hip_bf16.h>
#include <math.h>

// Problem constants (match reference)
#define TT   512
#define HH   1024
#define EE   16
#define KK   4
#define II   1024
#define F2I  2048   // 2*I
#define MOE_ALPHA 1.702f
#define MOE_LIMIT 7.0f

typedef __attribute__((ext_vector_type(16))) __bf16 v16bf;
typedef __attribute__((ext_vector_type(8)))  __bf16 bf16x8;
typedef __attribute__((ext_vector_type(8)))  float  v8f;
typedef __attribute__((ext_vector_type(4)))  float  f32x4;

union BF16Frag { v16bf v; bf16x8 h[2]; };

// ---------------- init: zero expert counters ----------------
__global__ void moe_init_kernel(int* __restrict__ cnt) {
    int i = threadIdx.x;
    if (i < EE) cnt[i] = 0;
}

// ---------------- router: logits -> top4 -> softmax -> lists ----------------
__global__ __launch_bounds__(256)
void moe_router_kernel(const float* __restrict__ x,
                       const float* __restrict__ rw,
                       const float* __restrict__ rb,
                       float* __restrict__ scores,
                       int* __restrict__ cnt,
                       int* __restrict__ lists) {
    int t = blockIdx.x * blockDim.x + threadIdx.x;
    if (t >= TT) return;

    float lg[EE];
#pragma unroll
    for (int e = 0; e < EE; ++e) lg[e] = rb[e];

    const float* xr = x + (size_t)t * HH;
    for (int h = 0; h < HH; ++h) {
        float xv = xr[h];
        const float* wr = rw + (size_t)h * EE;
#pragma unroll
        for (int e = 0; e < EE; ++e) lg[e] = fmaf(xv, wr[e], lg[e]);
    }

    int   idxs[KK];
    float vals[KK];
#pragma unroll
    for (int k = 0; k < KK; ++k) {
        float best = -1e30f; int bi = 0;
#pragma unroll
        for (int e = 0; e < EE; ++e) {
            if (lg[e] > best) { best = lg[e]; bi = e; }
        }
        idxs[k] = bi; vals[k] = best; lg[bi] = -1e30f;
    }

    float vmax = vals[0];
    float w[KK]; float s = 0.f;
#pragma unroll
    for (int k = 0; k < KK; ++k) { w[k] = __expf(vals[k] - vmax); s += w[k]; }
    float inv = 1.f / s;

#pragma unroll
    for (int k = 0; k < KK; ++k) {
        int e = idxs[k];
        scores[t * EE + e] = w[k] * inv;
        int pos = atomicAdd(&cnt[e], 1);
        lists[e * TT + pos] = t * KK + k;   // encoded (token, slot)
    }
}

// ---------------- expert: gathered 16-token tile, WMMA bf16 ----------------
__global__ __launch_bounds__(256)
void moe_expert_kernel(const float* __restrict__ x,
                       const float* __restrict__ gup,
                       const float* __restrict__ gub,
                       const float* __restrict__ dwn,
                       const float* __restrict__ dwb,
                       const float* __restrict__ scores,
                       const int* __restrict__ cnt,
                       const int* __restrict__ lists,
                       float* __restrict__ partial) {
    __shared__ __bf16 wLDS[512][32];      // 32 KB weight stage (bf16, [col][k])
    __shared__ __bf16 actLDS[16][1024];   // 32 KB activation tile

    const int e    = blockIdx.x;
    const int base = blockIdx.y * 16;
    const int n    = cnt[e];
    if (base >= n) return;

    const int tid   = threadIdx.x;
    const int wv    = tid >> 5;      // wave id 0..7
    const int lane  = tid & 31;
    const int half  = lane >> 4;     // K-subrange selector for A/B fragments
    const int mlane = lane & 15;

    // A-fragment row token (lane m owns matrix row m)
    int arow = base + mlane;
    int atok = 0;
    if (arow < n) atok = lists[e * TT + arow] >> 2;
    const float* xrow = x + (size_t)atok * HH;

    // ---------------- Phase A: act[16][1024] ----------------
    // Weight stage de-interleaves gate/up: LDS cols [0,256) = gate (even gu
    // cols), [256,512) = up (odd gu cols).  Each wave owns 2 gate tiles and
    // their 2 matching up tiles, so the activation is purely lane-local.
    for (int pass = 0; pass < 4; ++pass) {            // gu columns in 512-wide stripes
        const int colBase = pass * 512;
        v8f acc[4] = {};                              // [0..1]=gate tiles, [2..3]=up tiles
        for (int k0 = 0; k0 < HH; k0 += 32) {
            __syncthreads();
            // stage gate_up[k0:k0+32, colBase:colBase+512] transposed + de-interleaved
            const float* wbase = gup + ((size_t)e * HH + k0) * F2I + colBase;
#pragma unroll 8
            for (int p = 0; p < 32; ++p) {
                int idx = tid + p * 256;     // 0..8191
                int c   = idx & 511;         // memory column within stripe (coalesced)
                int kp  = idx >> 9;          // row pair 0..15
                int cl  = (c >> 1) | ((c & 1) << 8);   // de-interleave: gate->0..255, up->256..511
                float a = wbase[(size_t)(2 * kp)     * F2I + c];
                float b = wbase[(size_t)(2 * kp + 1) * F2I + c];
                wLDS[cl][2 * kp]     = (__bf16)a;
                wLDS[cl][2 * kp + 1] = (__bf16)b;
            }
            __syncthreads();
            // prefetch next chunk toward L2/L0 while this chunk computes
            if (k0 + 32 < HH) {
                const float* nb = wbase + (size_t)32 * F2I;
                int l0 = tid * 2;                       // 512 cachelines of 128B
                __builtin_prefetch(nb + (size_t)(l0 >> 4) * F2I + (l0 & 15) * 32, 0, 0);
                __builtin_prefetch(nb + (size_t)((l0 + 1) >> 4) * F2I + ((l0 + 1) & 15) * 32, 0, 0);
            }
            // A fragment: x[row, k0 + half*8 .. +8) and (k0+16+half*8 .. +8)
            BF16Frag af;
#pragma unroll
            for (int j = 0; j < 8; ++j) {
                af.v[j]     = (__bf16)xrow[k0 + half * 8 + j];
                af.v[j + 8] = (__bf16)xrow[k0 + 16 + half * 8 + j];
            }
            // preload all 4 B fragments, then issue WMMAs back-to-back
            BF16Frag bfr[4];
#pragma unroll
            for (int q = 0; q < 2; ++q) {
                int ccg = wv * 32 + q * 16 + mlane;          // gate tile column
                int ccu = 256 + ccg;                         // matching up tile column
                bfr[q].h[0]     = *(const bf16x8*)&wLDS[ccg][half * 8];
                bfr[q].h[1]     = *(const bf16x8*)&wLDS[ccg][16 + half * 8];
                bfr[2 + q].h[0] = *(const bf16x8*)&wLDS[ccu][half * 8];
                bfr[2 + q].h[1] = *(const bf16x8*)&wLDS[ccu][16 + half * 8];
            }
#pragma unroll
            for (int q = 0; q < 4; ++q) {
                acc[q] = __builtin_amdgcn_wmma_f32_16x16x32_bf16(
                    false, af.v, false, bfr[q].v, (short)0, acc[q], false, false);
            }
        }
        // fused bias + clamp + GLU activation: fully lane-local, no shuffles
#pragma unroll
        for (int q = 0; q < 2; ++q) {
            int icol = (colBase >> 1) + wv * 32 + q * 16 + mlane;   // act column i
            float gbias = gub[(size_t)e * F2I + 2 * icol];
            float ubias = gub[(size_t)e * F2I + 2 * icol + 1];
#pragma unroll
            for (int r = 0; r < 8; ++r) {
                float gate = acc[q][r] + gbias;
                float up   = acc[2 + q][r] + ubias;
                gate = fminf(gate, MOE_LIMIT);
                up   = fminf(fmaxf(up, -MOE_LIMIT), MOE_LIMIT);
                float glu = gate * __builtin_amdgcn_rcpf(1.f + __expf(-MOE_ALPHA * gate));
                actLDS[r + 8 * half][icol] = (__bf16)((up + 1.f) * glu);
            }
        }
    }
    __syncthreads();

    // epilogue row -> (encoded slot, router weight)
    int   benc[8];
    float bwt[8];
    int   bval[8];
#pragma unroll
    for (int r = 0; r < 8; ++r) {
        int row = base + r + 8 * half;
        int enc = 0; float wgt = 0.f; int valid = 0;
        if (row < n) {
            enc = lists[e * TT + row];
            wgt = scores[(enc >> 2) * EE + e];
            valid = 1;
        }
        benc[r] = enc; bwt[r] = wgt; bval[r] = valid;
    }

    // ---------------- Phase B: out tile = act @ down + bias, scaled ----------------
    for (int pass = 0; pass < 2; ++pass) {            // H columns in 512-wide stripes
        const int hBase = pass * 512;
        v8f acc[4] = {};
        for (int k0 = 0; k0 < II; k0 += 32) {
            __syncthreads();
            const float* wbase = dwn + ((size_t)e * II + k0) * HH + hBase;
#pragma unroll 8
            for (int p = 0; p < 32; ++p) {
                int idx = tid + p * 256;
                int c   = idx & 511;
                int kp  = idx >> 9;
                float a = wbase[(size_t)(2 * kp)     * HH + c];
                float b = wbase[(size_t)(2 * kp + 1) * HH + c];
                wLDS[c][2 * kp]     = (__bf16)a;
                wLDS[c][2 * kp + 1] = (__bf16)b;
            }
            __syncthreads();
            if (k0 + 32 < II) {
                const float* nb = wbase + (size_t)32 * HH;
                int l0 = tid;                           // 256 cachelines of 128B
                __builtin_prefetch(nb + (size_t)(l0 >> 3) * HH + (l0 & 7) * 32, 0, 0);
            }
            BF16Frag af;
            af.h[0] = *(const bf16x8*)&actLDS[mlane][k0 + half * 8];
            af.h[1] = *(const bf16x8*)&actLDS[mlane][k0 + 16 + half * 8];
            BF16Frag bfr[4];
#pragma unroll
            for (int nt = 0; nt < 4; ++nt) {
                int cc = wv * 64 + nt * 16 + mlane;
                bfr[nt].h[0] = *(const bf16x8*)&wLDS[cc][half * 8];
                bfr[nt].h[1] = *(const bf16x8*)&wLDS[cc][16 + half * 8];
            }
#pragma unroll
            for (int nt = 0; nt < 4; ++nt) {
                acc[nt] = __builtin_amdgcn_wmma_f32_16x16x32_bf16(
                    false, af.v, false, bfr[nt].v, (short)0, acc[nt], false, false);
            }
        }
#pragma unroll
        for (int nt = 0; nt < 4; ++nt) {
            int hcol = hBase + wv * 64 + nt * 16 + mlane;
            float dbias = dwb[(size_t)e * HH + hcol];
#pragma unroll
            for (int r = 0; r < 8; ++r) {
                if (bval[r]) {
                    float contrib = bwt[r] * (acc[nt][r] + dbias);
                    partial[(size_t)benc[r] * HH + hcol] = contrib;
                }
            }
        }
    }
}

// ---------------- combine: fixed-order sum over K slots (vectorized) ----------------
__global__ __launch_bounds__(256)
void moe_combine_kernel(const float* __restrict__ partial, float* __restrict__ out) {
    int idx = blockIdx.x * blockDim.x + threadIdx.x;   // one float4 per thread
    if (idx >= (TT * HH) / 4) return;
    int t  = idx / (HH / 4);
    int h4 = idx % (HH / 4);
    f32x4 s = {};
#pragma unroll
    for (int k = 0; k < KK; ++k) {
        f32x4 v = *(const f32x4*)&partial[((size_t)t * KK + k) * HH + h4 * 4];
        s += v;
    }
    *(f32x4*)&out[(size_t)t * HH + h4 * 4] = s;
}

extern "C" void kernel_launch(void* const* d_in, const int* in_sizes, int n_in,
                              void* d_out, int out_size, void* d_ws, size_t ws_size,
                              hipStream_t stream) {
    (void)in_sizes; (void)n_in; (void)out_size; (void)ws_size;
    const float* x   = (const float*)d_in[0];  // [T,H]
    const float* rw  = (const float*)d_in[1];  // [H,E]
    const float* rb  = (const float*)d_in[2];  // [E]
    const float* gup = (const float*)d_in[3];  // [E,H,2I]
    const float* gub = (const float*)d_in[4];  // [E,2I]
    const float* dwn = (const float*)d_in[5];  // [E,I,H]
    const float* dwb = (const float*)d_in[6];  // [E,H]
    float* out = (float*)d_out;                // [T,H]

    char* ws = (char*)d_ws;
    float* scores  = (float*)ws;                                   // T*E f32 = 32 KB
    int*   cnt     = (int*)(ws + (size_t)TT * EE * 4);             // E ints
    int*   lists   = (int*)(ws + (size_t)TT * EE * 4 + 64);        // E*T ints = 32 KB
    float* partial = (float*)(ws + (size_t)TT * EE * 4 + 64 + (size_t)EE * TT * 4); // T*K*H f32 = 8 MB

    moe_init_kernel<<<1, 64, 0, stream>>>(cnt);
    moe_router_kernel<<<(TT + 255) / 256, 256, 0, stream>>>(x, rw, rb, scores, cnt, lists);
    moe_expert_kernel<<<dim3(EE, TT / 16), dim3(256), 0, stream>>>(
        x, gup, gub, dwn, dwb, scores, cnt, lists, partial);
    moe_combine_kernel<<<(TT * HH / 4 + 255) / 256, 256, 0, stream>>>(partial, out);
}
